// CausalSelfAttention_85890755986095
// MI455X (gfx1250) — compile-verified, matrix operand order checked by reading, not run on hardware
//
#include <hip/hip_runtime.h>
#include <stdint.h>

// ---------------- problem constants ----------------
#define B_   4
#define T_   2048
#define C_   1024
#define H_   16
#define D_   64
#define C3_  3072

typedef __attribute__((ext_vector_type(16))) _Float16 v16h;
typedef __attribute__((ext_vector_type(8)))  _Float16 v8h;
typedef __attribute__((ext_vector_type(8)))  float    v8f;
typedef __attribute__((ext_vector_type(4)))  unsigned int u32x4;
typedef __attribute__((ext_vector_type(4)))  int      i32x4;
typedef __attribute__((ext_vector_type(8)))  int      i32x8;

// ---------------- CDNA5 helpers ----------------

// LDS byte address of a shared-memory pointer (flat addr low 32 bits == LDS offset)
__device__ __forceinline__ uint32_t lds_off(const void* p) {
    return (uint32_t)(uintptr_t)p;
}

// Async 16-byte global->LDS copy (GVS mode: SGPR64 base + VGPR32 byte offset).
__device__ __forceinline__ void async_cp16(void* lds_dst, const void* gbase, uint32_t gbyteoff) {
    asm volatile("global_load_async_to_lds_b128 %0, %1, %2"
                 :: "v"(lds_off(lds_dst)), "v"(gbyteoff), "s"(gbase)
                 : "memory");
}
__device__ __forceinline__ void wait_async() {
    asm volatile("s_wait_asynccnt 0" ::: "memory");
}
__device__ __forceinline__ void wait_tensor() {
    __builtin_amdgcn_s_wait_tensorcnt(0);
}

// Tensor Data Mover: 2D tile load (tile_h rows x tile_w f16 elems, row stride
// `stride_elems`) from global `gptr` (tile start) into LDS at byte `lds_byte`.
// Issued once per executing wave; tracked by TENSORcnt.
__device__ __forceinline__ void tdm_load_2d(uint32_t lds_byte, const void* gptr,
                                            uint32_t stride_elems, uint32_t rows_total,
                                            uint32_t tile_w, uint32_t tile_h) {
    uint64_t ga = (uint64_t)(uintptr_t)gptr;
    u32x4 g0;
    g0[0] = 1u;                                            // count=1 (valid user D#)
    g0[1] = lds_byte;                                      // lds_addr
    g0[2] = (uint32_t)ga;                                  // global_addr[31:0]
    g0[3] = (uint32_t)((ga >> 32) & 0x01FFFFFFu)           // global_addr[56:32]
            | (2u << 30);                                  // type=2 ("image")
    i32x8 g1;
    g1[0] = 0x00010000;                                    // data_size=2B, wg_mask=0
    g1[1] = (int)((stride_elems & 0xFFFFu) << 16);         // tensor_dim0[15:0]
    g1[2] = (int)((stride_elems >> 16) | ((rows_total & 0xFFFFu) << 16)); // dim0 hi | dim1 lo
    g1[3] = (int)((rows_total >> 16) | (tile_w << 16));    // dim1 hi | tile_dim0
    g1[4] = (int)tile_h;                                   // tile_dim1 (tile_dim2=0)
    g1[5] = (int)stride_elems;                             // tensor_dim0_stride[31:0]
    g1[6] = 0;                                             // stride0 hi | stride1 lo
    g1[7] = 0;
    i32x4 z4 = {0, 0, 0, 0};
#if __has_include(<hip/amd_detail/amd_gfx1250_TDM.h>)
    i32x8 z8 = {0, 0, 0, 0, 0, 0, 0, 0};
    __builtin_amdgcn_tensor_load_to_lds(g0, g1, z4, z4, z8, 0);
#else
    __builtin_amdgcn_tensor_load_to_lds(g0, g1, z4, z4, 0);
#endif
}

// Load a 16x32 f16 A/B fragment (ISA 7.12.2 layout) from LDS or global.
__device__ __forceinline__ v16h load_frag(const _Float16* t, int ld) {
    int lane = threadIdx.x & 31;
    int row  = lane & 15;
    int kh   = (lane >> 4) << 3;          // 0 or 8
    const _Float16* p = t + row * ld + kh;
    v8h lo = *(const v8h*)(p);
    v8h hi = *(const v8h*)(p + 16);
    return __builtin_shufflevector(lo, hi, 0,1,2,3,4,5,6,7,8,9,10,11,12,13,14,15);
}

__device__ __forceinline__ v8f wmma32(v16h a, v16h b, v8f c) {
    return __builtin_amdgcn_wmma_f32_16x16x32_f16(false, a, false, b, (short)0, c, false, false);
}

__device__ __forceinline__ v8f zero8() {
    v8f z = {0.f,0.f,0.f,0.f,0.f,0.f,0.f,0.f};
    return z;
}

// ---------------- kernel 0: f32 -> f16 convert ----------------
__global__ void cvt_f16_kernel(const float* __restrict__ src, _Float16* __restrict__ dst, int n) {
    int i = blockIdx.x * blockDim.x + threadIdx.x;
    int stride = gridDim.x * blockDim.x;
    for (; i < n; i += stride) dst[i] = (_Float16)src[i];
}

// ---------------- kernel 1/4: C = A(MxK) * Bw(NxK)^T + bias ----------------
// 64x64 tile per 128-thread block; TDM-staged LDS tiles, double buffered.
__global__ void __launch_bounds__(128)
gemm_f16f32_kernel(const _Float16* __restrict__ A, const _Float16* __restrict__ Bw,
                   const float* __restrict__ bias, float* __restrict__ Cout,
                   int N, int K) {
    __shared__ __align__(16) _Float16 At[2][64 * 32];
    __shared__ __align__(16) _Float16 Bt[2][64 * 32];

    const int n0   = blockIdx.x * 64;
    const int m0   = blockIdx.y * 64;
    const int wave = threadIdx.x >> 5;
    const int lane = threadIdx.x & 31;
    const int g    = lane >> 4;
    const int cc   = lane & 15;

    // wave0 DMAs the A tile, wave1 DMAs the B tile (TDM, TENSORcnt-tracked)
    auto stage = [&](int buf, int k0) {
        if (wave == 0)
            tdm_load_2d(lds_off(&At[buf][0]), A + (size_t)m0 * K + k0, K, 64, 32, 64);
        else if (wave == 1)
            tdm_load_2d(lds_off(&Bt[buf][0]), Bw + (size_t)n0 * K + k0, K, 64, 32, 64);
    };

    v8f acc[4] = { zero8(), zero8(), zero8(), zero8() };

    stage(0, 0);
    if (wave < 2) wait_tensor();
    __syncthreads();

    const int nk = K >> 5;
    for (int kc = 0; kc < nk; ++kc) {
        const int buf = kc & 1;
        if (kc + 1 < nk) stage(1 - buf, (kc + 1) << 5);   // prefetch next tile

        v16h a = load_frag(&At[buf][wave * 16 * 32], 32);
        v16h b[4];
        #pragma unroll
        for (int n = 0; n < 4; ++n) b[n] = load_frag(&Bt[buf][n * 16 * 32], 32);
        #pragma unroll
        for (int n = 0; n < 4; ++n) acc[n] = wmma32(a, b[n], acc[n]);

        if (wave < 2) wait_tensor();
        __syncthreads();
    }

    #pragma unroll
    for (int n = 0; n < 4; ++n) {
        int col = n0 + n * 16 + cc;
        float bs = bias[col];
        #pragma unroll
        for (int r = 0; r < 8; ++r) {
            int row = m0 + wave * 16 + g * 8 + r;
            Cout[(size_t)row * N + col] = acc[n][r] + bs;
        }
    }
}

// ---------------- kernel 2: blend value, rms-norm + rotary q/k ----------------
__global__ void __launch_bounds__(128)
epilogue_kernel(const float* __restrict__ qkv, const float* __restrict__ v1,
                const float* __restrict__ lambp, float* __restrict__ value_out,
                _Float16* __restrict__ Qh, _Float16* __restrict__ Kh,
                _Float16* __restrict__ Vt) {
    const int wave = threadIdx.x >> 5;
    const int lane = threadIdx.x & 31;
    const int rowid = blockIdx.x * 4 + wave;     // (b,h,t) flattened
    const int t  = rowid & (T_ - 1);
    const int bh = rowid >> 11;                  // b*H + h
    const int h  = bh & (H_ - 1);
    const int b  = bh >> 4;

    const float lam = lambp[0];
    const size_t base  = ((size_t)(b * T_ + t)) * C3_ + h * D_;
    const size_t vbase = ((size_t)bh * T_ + t) * D_;
    const int d0 = lane, d1 = lane + 32;

    float va = (1.f - lam) * qkv[base + 2 * C_ + d0] + lam * v1[vbase + d0];
    float vb = (1.f - lam) * qkv[base + 2 * C_ + d1] + lam * v1[vbase + d1];
    value_out[vbase + d0] = va;
    value_out[vbase + d1] = vb;
    Vt[((size_t)bh * D_ + d0) * T_ + t] = (_Float16)va;
    Vt[((size_t)bh * D_ + d1) * T_ + t] = (_Float16)vb;

    float inv = __powf(10000.f, -(float)lane * (1.f / 32.f));
    float ang = (float)t * inv;
    float cs = __cosf(ang), sn = __sinf(ang);

    {
        float x1 = qkv[base + d0], x2 = qkv[base + d1];
        float ss = x1 * x1 + x2 * x2;
        #pragma unroll
        for (int off = 16; off >= 1; off >>= 1) ss += __shfl_xor(ss, off, 32);
        float r = rsqrtf(ss * (1.f / 64.f) + 1e-6f);
        x1 *= r; x2 *= r;
        Qh[vbase + d0] = (_Float16)( x1 * cs + x2 * sn);
        Qh[vbase + d1] = (_Float16)(-x1 * sn + x2 * cs);
    }
    {
        float x1 = qkv[base + C_ + d0], x2 = qkv[base + C_ + d1];
        float ss = x1 * x1 + x2 * x2;
        #pragma unroll
        for (int off = 16; off >= 1; off >>= 1) ss += __shfl_xor(ss, off, 32);
        float r = rsqrtf(ss * (1.f / 64.f) + 1e-6f);
        x1 *= r; x2 *= r;
        Kh[vbase + d0] = (_Float16)( x1 * cs + x2 * sn);
        Kh[vbase + d1] = (_Float16)(-x1 * sn + x2 * cs);
    }
}

// ---------------- kernel 3: causal flash attention ----------------
// Block: (bh, qtile). 4 waves x 16 query rows; async-LDS staged K/V^T tiles,
// double buffered so the copy of tile kt+1 overlaps compute on tile kt.
__global__ void __launch_bounds__(128)
attn_kernel(const _Float16* __restrict__ Qh, const _Float16* __restrict__ Kh,
            const _Float16* __restrict__ Vt, _Float16* __restrict__ Oh) {
    __shared__ __align__(16) _Float16 Kt[2][64 * 64];    // [key][d]
    __shared__ __align__(16) _Float16 Vtt[2][64 * 64];   // [d][key]
    __shared__ __align__(16) _Float16 Pt[4 * 16 * 64];   // per-wave P staging

    const int bh   = blockIdx.x;          // 0..63
    const int qt   = blockIdx.y;          // 0..31
    const int tid  = threadIdx.x;
    const int wave = tid >> 5;
    const int lane = tid & 31;
    const int g    = lane >> 4;
    const int cc   = lane & 15;
    const int t0   = qt * 64 + wave * 16;

    auto stage = [&](int buf, int kt) {
        #pragma unroll
        for (int i = 0; i < 4; ++i) {
            int cid = tid + i * 128;
            int row = cid >> 3;
            int cj  = cid & 7;
            async_cp16(&Kt[buf][row * 64 + cj * 8], Kh,
                       (uint32_t)((((size_t)bh * T_ + kt * 64 + row) * 64 + cj * 8) * 2));
            async_cp16(&Vtt[buf][row * 64 + cj * 8], Vt,
                       (uint32_t)((((size_t)bh * D_ + row) * T_ + kt * 64 + cj * 8) * 2));
        }
    };

    const _Float16* Qbase = Qh + ((size_t)bh * T_ + t0) * D_;
    v16h qa0 = load_frag(Qbase, 64);
    v16h qa1 = load_frag(Qbase + 32, 64);

    v8f ov[4] = { zero8(), zero8(), zero8(), zero8() };
    float m[8], l[8];
    #pragma unroll
    for (int r = 0; r < 8; ++r) { m[r] = -1e30f; l[r] = 0.f; }
    const float scale = 0.125f;           // 1/sqrt(64)

    stage(0, 0);
    wait_async();
    __syncthreads();

    for (int kt = 0; kt <= qt; ++kt) {
        const int buf = kt & 1;
        if (kt < qt) stage(1 - buf, kt + 1);              // prefetch next K/V tile

        // ---- S = Q K^T: preload all 8 B-fragments, then chain 8 WMMAs
        v16h kb[8];
        #pragma unroll
        for (int n = 0; n < 4; ++n) {
            kb[2 * n]     = load_frag(&Kt[buf][n * 16 * 64], 64);
            kb[2 * n + 1] = load_frag(&Kt[buf][n * 16 * 64 + 32], 64);
        }
        float p[4][8];
        float rowmax[8];
        #pragma unroll
        for (int r = 0; r < 8; ++r) rowmax[r] = -1e30f;
        #pragma unroll
        for (int n = 0; n < 4; ++n) {
            v8f s = wmma32(qa1, kb[2 * n + 1], wmma32(qa0, kb[2 * n], zero8()));
            int kidx = kt * 64 + n * 16 + cc;
            #pragma unroll
            for (int r = 0; r < 8; ++r) {
                int qidx = t0 + g * 8 + r;
                float val = (kidx <= qidx) ? s[r] * scale : -1e30f;
                p[n][r] = val;
                rowmax[r] = fmaxf(rowmax[r], val);
            }
        }

        // ---- online softmax (row stats live in the 16-lane group)
        float alpha[8];
        #pragma unroll
        for (int r = 0; r < 8; ++r) {
            float mx = rowmax[r];
            #pragma unroll
            for (int off = 8; off >= 1; off >>= 1) mx = fmaxf(mx, __shfl_xor(mx, off, 32));
            float mn = fmaxf(m[r], mx);
            alpha[r] = __expf(m[r] - mn);
            float rs = 0.f;
            #pragma unroll
            for (int n = 0; n < 4; ++n) { p[n][r] = __expf(p[n][r] - mn); rs += p[n][r]; }
            #pragma unroll
            for (int off = 8; off >= 1; off >>= 1) rs += __shfl_xor(rs, off, 32);
            l[r] = l[r] * alpha[r] + rs;
            m[r] = mn;
        }
        #pragma unroll
        for (int n = 0; n < 4; ++n)
            #pragma unroll
            for (int r = 0; r < 8; ++r) ov[n][r] *= alpha[r];

        // ---- stage P through LDS: C/D layout -> A layout (wave-private tile)
        _Float16* pw = Pt + wave * 16 * 64;
        #pragma unroll
        for (int n = 0; n < 4; ++n)
            #pragma unroll
            for (int r = 0; r < 8; ++r)
                pw[(g * 8 + r) * 64 + n * 16 + cc] = (_Float16)p[n][r];

        v16h pa0 = load_frag(pw, 64);
        v16h pa1 = load_frag(pw + 32, 64);
        v16h vb[8];
        #pragma unroll
        for (int n = 0; n < 4; ++n) {
            vb[2 * n]     = load_frag(&Vtt[buf][n * 16 * 64], 64);
            vb[2 * n + 1] = load_frag(&Vtt[buf][n * 16 * 64 + 32], 64);
        }
        #pragma unroll
        for (int n = 0; n < 4; ++n) {
            ov[n] = wmma32(pa0, vb[2 * n], ov[n]);
            ov[n] = wmma32(pa1, vb[2 * n + 1], ov[n]);
        }

        wait_async();          // drain prefetch of next tile before buffer swap
        __syncthreads();
    }

    // ---- normalize, write O as f16 in (b,t,h*D+d) layout for the proj GEMM
    const int b = bh >> 4, h = bh & 15;
    #pragma unroll
    for (int n = 0; n < 4; ++n) {
        #pragma unroll
        for (int r = 0; r < 8; ++r) {
            int t = t0 + g * 8 + r;
            Oh[((size_t)(b * T_ + t)) * C_ + h * D_ + n * 16 + cc] =
                (_Float16)(ov[n][r] / l[r]);
        }
    }
}

// ---------------- launcher ----------------
extern "C" void kernel_launch(void* const* d_in, const int* in_sizes, int n_in,
                              void* d_out, int out_size, void* d_ws, size_t ws_size,
                              hipStream_t stream) {
    const float* x      = (const float*)d_in[0];
    const float* v1     = (const float*)d_in[1];
    const float* W_qkv  = (const float*)d_in[2];
    const float* b_qkv  = (const float*)d_in[3];
    const float* W_proj = (const float*)d_in[4];
    const float* b_proj = (const float*)d_in[5];
    const float* lamb   = (const float*)d_in[6];

    float* out       = (float*)d_out;                          // (B,T,C) f32
    float* value_out = out + (size_t)B_ * T_ * C_;             // (B,H,T,D) f32

    char* ws = (char*)d_ws;
    size_t off = 0;
    auto alloc = [&](size_t bytes) -> void* {
        void* p = ws + off;
        off += (bytes + 255) & ~(size_t)255;
        return p;
    };
    _Float16* xh  = (_Float16*)alloc((size_t)B_ * T_ * C_ * 2);
    _Float16* Wqh = (_Float16*)alloc((size_t)C3_ * C_ * 2);
    _Float16* Wph = (_Float16*)alloc((size_t)C_ * C_ * 2);
    float*    qkv = (float*)   alloc((size_t)B_ * T_ * C3_ * 4);
    _Float16* Qh  = (_Float16*)alloc((size_t)B_ * H_ * T_ * D_ * 2);
    _Float16* Kh  = (_Float16*)alloc((size_t)B_ * H_ * T_ * D_ * 2);
    _Float16* Vt  = (_Float16*)alloc((size_t)B_ * H_ * T_ * D_ * 2);
    _Float16* Oh  = (_Float16*)alloc((size_t)B_ * T_ * C_ * 2);
    if (off > ws_size) return;  // workspace too small; bail deterministically

    // 0) convert inputs to f16
    cvt_f16_kernel<<<1024, 256, 0, stream>>>(x, xh, B_ * T_ * C_);
    cvt_f16_kernel<<<512, 256, 0, stream>>>(W_qkv, Wqh, C3_ * C_);
    cvt_f16_kernel<<<256, 256, 0, stream>>>(W_proj, Wph, C_ * C_);

    // 1) qkv = x @ W_qkv^T + b_qkv   (f32 result in ws)
    gemm_f16f32_kernel<<<dim3(C3_ / 64, (B_ * T_) / 64), 128, 0, stream>>>(
        xh, Wqh, b_qkv, qkv, C3_, C_);

    // 2) value blend + rms-norm + rotary -> Qh, Kh, Vt(f16), value_out(f32)
    epilogue_kernel<<<(B_ * H_ * T_) / 4, 128, 0, stream>>>(
        qkv, v1, lamb, value_out, Qh, Kh, Vt);

    // 3) causal flash attention -> Oh (f16, (B,T,C) layout)
    attn_kernel<<<dim3(B_ * H_, T_ / 64), 128, 0, stream>>>(Qh, Kh, Vt, Oh);

    // 4) out = O @ W_proj^T + b_proj
    gemm_f16f32_kernel<<<dim3(C_ / 64, (B_ * T_) / 64), 128, 0, stream>>>(
        Oh, Wph, b_proj, out, C_, C_);
}
